// LRU_61014305407393
// MI455X (gfx1250) — matrix operand order repaired and checked by reading, not compiled, not run on hardware
//
#include <hip/hip_runtime.h>

typedef __attribute__((ext_vector_type(16))) __bf16        v16bf;
typedef __attribute__((ext_vector_type(8)))  float         v8f;
typedef __attribute__((ext_vector_type(8)))  unsigned int  v8u;

namespace {
constexpr int kB    = 16;      // batch
constexpr int kL0   = 200;     // seq len
constexpr int kLp   = 256;     // padded seq len (front pad 56)
constexpr int kD    = 256;     // model dim
constexpr int kNB   = 2;       // blocks
constexpr int kV    = 8001;    // vocab
constexpr int kVp   = 8016;    // vocab padded to /16
constexpr int kDL   = 768;
constexpr int kDI   = 512;
constexpr int kDLI  = 1280;
constexpr float kEps = 1e-5f;

constexpr size_t SZ_TAB   = (size_t)kVp * kDLI;   // bf16 concat tables (padded)
constexpr size_t SZ_PROJW = (size_t)kD  * kDLI;   // bf16 proj weight
constexpr size_t SZ_WEMB  = (size_t)kVp * kD;     // f32 + bf16 copies
constexpr size_t SZ_ACT   = (size_t)kB * kLp * kD;// 1,048,576
constexpr size_t SZ_W     = (size_t)kNB * kD * kD;
constexpr size_t SZ_XO    = (size_t)kB * kL0 * kD;
} // namespace

__device__ __forceinline__ unsigned short f2bf(float f) {
  unsigned int u = __float_as_uint(f);
  u += 0x7fffu + ((u >> 16) & 1u);          // round-to-nearest-even
  return (unsigned short)(u >> 16);
}
__device__ __forceinline__ unsigned int ld32(const unsigned short* p) {
  return *(const unsigned int*)p;           // 2 packed bf16 (4B aligned: even k, even ld)
}

// ---------------------------------------------------------------------------
// Generic bf16 WMMA GEMM:  C[m,n] = alpha * (A @ B^T)[m,n] * colScale[n]
//                                   + bias[n] + addm[m,n]
// A: M x K (row-major, bf16, lda), B: N x K (row-major, bf16, ldb).
// One wave = one 16x16 C tile; fragment layouts per CDNA5 ISA 7.12.2 (wave32).
// ---------------------------------------------------------------------------
__global__ __launch_bounds__(128) void k_gemm_bf16(
    const unsigned short* __restrict__ A, int lda,
    const unsigned short* __restrict__ B, int ldb,
    float* __restrict__ C, int ldc,
    int mtiles, int Nstore, int K, float alpha,
    const float* __restrict__ bias,
    const float* __restrict__ colScale,
    const float* __restrict__ addm, int ldadd)
{
  const int wave  = threadIdx.x >> 5;
  const int lane  = threadIdx.x & 31;
  const int mtile = blockIdx.y * 4 + wave;
  if (mtile >= mtiles) return;              // wave-uniform: EXEC stays all-1s for WMMA
  const int ntile = blockIdx.x;
  const int m0 = mtile * 16, n0 = ntile * 16;
  const int rlo   = lane & 15;
  const int khalf = lane >> 4;

  const unsigned short* Arow = A + (size_t)(m0 + rlo) * lda;
  const unsigned short* Brow = B + (size_t)(n0 + rlo) * ldb;

  v8f c = {0.f, 0.f, 0.f, 0.f, 0.f, 0.f, 0.f, 0.f};

  for (int k0 = 0; k0 < K; k0 += 32) {
    // A 16x32 bf16 fragment: lanes 0-15 rows M, K 0..7 (V0-3) / 16..23 (V4-7);
    // lanes 16-31 same rows, K 8..15 / 24..31.
    const int ka = k0 + khalf * 8;
    v8u au;
    au[0] = ld32(Arow + ka + 0);
    au[1] = ld32(Arow + ka + 2);
    au[2] = ld32(Arow + ka + 4);
    au[3] = ld32(Arow + ka + 6);
    au[4] = ld32(Arow + ka + 16);
    au[5] = ld32(Arow + ka + 18);
    au[6] = ld32(Arow + ka + 20);
    au[7] = ld32(Arow + ka + 22);
    // B 32x16 bf16 fragment: lanes 0-15 col N, K 0..7 / 16..23; lanes 16-31 K 8..15 / 24..31.
    const int kb = k0 + khalf * 8;
    v8u bu;
#pragma unroll
    for (int v = 0; v < 8; ++v) bu[v] = ld32(Brow + kb + 2 * v + (v >= 4 ? 8 : 0));

    __builtin_prefetch(Arow + ka + 32, 0, 1);   // global_prefetch_b8
    __builtin_prefetch(Brow + kb + 32, 0, 1);

    v16bf a = __builtin_bit_cast(v16bf, au);
    v16bf b = __builtin_bit_cast(v16bf, bu);
    c = __builtin_amdgcn_wmma_f32_16x16x32_bf16(
        /*neg_a=*/false, a, /*neg_b=*/false, b,
        /*c_mod=*/(short)0, c, /*reuse_a=*/false, /*reuse_b=*/false);
  }

  const int n = n0 + rlo;
  const float cs = colScale ? colScale[n] : 1.0f;
  const float bs = bias ? bias[n] : 0.0f;
#pragma unroll
  for (int r = 0; r < 8; ++r) {             // C layout: M = r + 8*khalf, N = lane&15
    const int m = m0 + khalf * 8 + r;
    if (n < Nstore) {
      float val = c[r] * alpha * cs + bs;
      if (addm) val += addm[(size_t)m * ldadd + n];
      C[(size_t)m * ldc + n] = val;
    }
  }
}

// --------------------------- elementwise helpers ---------------------------
__global__ void k_f32_to_bf16(const float* __restrict__ in,
                              unsigned short* __restrict__ out, size_t n) {
  size_t i = (size_t)blockIdx.x * blockDim.x + threadIdx.x;
  if (i < n) out[i] = f2bf(in[i]);
}

__global__ void k_exp(const float* __restrict__ in, float* __restrict__ out, int n) {
  int i = blockIdx.x * blockDim.x + threadIdx.x;
  if (i < n) out[i] = __expf(in[i]);
}

__global__ void k_silu_mul(const float* __restrict__ g, const float* __restrict__ x1,
                           float* __restrict__ out, size_t n) {
  size_t i = (size_t)blockIdx.x * blockDim.x + threadIdx.x;
  if (i < n) {
    float gv = g[i];
    out[i] = gv / (1.0f + __expf(-gv)) * x1[i];
  }
}

// concat [lang|img] -> bf16, rows >= V zero-padded
__global__ void k_build_tables(const float* __restrict__ lang,
                               const float* __restrict__ img,
                               unsigned short* __restrict__ out) {
  size_t i = (size_t)blockIdx.x * blockDim.x + threadIdx.x;
  if (i >= SZ_TAB) return;
  int k = (int)(i % kDLI);
  int v = (int)(i / kDLI);
  float f = 0.0f;
  if (v < kV) f = (k < kDL) ? lang[(size_t)v * kDL + k] : img[(size_t)v * kDI + (k - kDL)];
  out[i] = f2bf(f);
}

// h[b, t, :] = (t >= pad) ? W_emb[x[b, t-pad], :] : 0 ; mask likewise
__global__ void k_gather_embed(const int* __restrict__ x,
                               const float* __restrict__ Wemb,
                               float* __restrict__ h, int* __restrict__ mask) {
  size_t i = (size_t)blockIdx.x * blockDim.x + threadIdx.x;
  if (i >= SZ_ACT) return;
  int d = (int)(i % kD);
  size_t r = i / kD;
  int t = (int)(r % kLp);
  int b = (int)(r / kLp);
  const int pad = kLp - kL0;
  float v = 0.0f;
  int mk = 0;
  if (t >= pad) {
    int tok = x[b * kL0 + (t - pad)];
    v = Wemb[(size_t)tok * kD + d];
    mk = (tok > 0) ? 1 : 0;
  }
  h[i] = v;
  if (d == 0) mask[b * kLp + t] = mk;
}

// xo_bf16[b*L0+l, :] = bf16( h[b, pad+l, :] )
__global__ void k_gather_xo(const float* __restrict__ h,
                            unsigned short* __restrict__ xo) {
  size_t i = (size_t)blockIdx.x * blockDim.x + threadIdx.x;
  if (i >= SZ_XO) return;
  int d = (int)(i % kD);
  size_t r = i / kD;
  int l = (int)(r % kL0);
  int b = (int)(r / kL0);
  xo[i] = f2bf(h[((size_t)b * kLp + (kLp - kL0) + l) * kD + d]);
}

// Matryoshka LayerNorm over segments [0,64),[64,128),[128,256); one row/block
__global__ void k_mrl_ln(const float* __restrict__ x, const float* __restrict__ g,
                         const float* __restrict__ bta, float* __restrict__ out) {
  const int row = blockIdx.x;
  const int t = threadIdx.x;
  __shared__ float sv[kD];
  __shared__ float stats[6];
  float v = x[(size_t)row * kD + t];
  sv[t] = v;
  __syncthreads();
  if (t < 3) {
    const int s0 = (t == 0) ? 0 : ((t == 1) ? 64 : 128);
    const int e0 = (t == 0) ? 64 : ((t == 1) ? 128 : 256);
    const float inv = 1.0f / (float)(e0 - s0);
    float mu = 0.0f;
    for (int i = s0; i < e0; ++i) mu += sv[i];
    mu *= inv;
    float var = 0.0f;
    for (int i = s0; i < e0; ++i) { float d = sv[i] - mu; var += d * d; }
    var *= inv;
    stats[t] = mu;
    stats[3 + t] = rsqrtf(var + kEps);
  }
  __syncthreads();
  const int seg = (t < 64) ? 0 : ((t < 128) ? 1 : 2);
  out[(size_t)row * kD + t] = (v - stats[seg]) * stats[3 + seg] * g[t] + bta[t];
}

// log-depth masked LRU scan; one block per (d, b); 256 threads over time.
// lambda^k = exp(-k*nu) * (cos(k*theta) + i sin(k*theta))
__global__ void k_lru_scan(float* __restrict__ hr_g, float* __restrict__ hi_g,
                           const int* __restrict__ mask,
                           const float* __restrict__ nu,
                           const float* __restrict__ theta) {
  const int d = blockIdx.x, b = blockIdx.y, t = threadIdx.x;
  __shared__ float sr[kLp], si[kLp];
  const size_t base = ((size_t)b * kLp + t) * kD + d;
  sr[t] = hr_g[base];
  si[t] = hi_g[base];
  const float nu_d = nu[d], th_d = theta[d];
  const int* mrow = mask + b * kLp;
  for (int l = 2; l <= kLp; l <<= 1) {
    __syncthreads();
    const int pos = t & (l - 1);
    const int half = l >> 1;
    if (pos >= half) {
      const int bnd = t - pos + half - 1;      // first-half element: not written this level
      if (mrow[bnd]) {
        const float k = (float)(pos - half + 1);
        const float s = __expf(-nu_d * k);
        const float wr = s * __cosf(th_d * k);
        const float wi = s * __sinf(th_d * k);
        const float br = sr[bnd], bi = si[bnd];
        sr[t] += wr * br - wi * bi;
        si[t] += wr * bi + wi * br;
      }
    }
  }
  __syncthreads();
  hr_g[base] = sr[t];
  hi_g[base] = si[t];
}

// ---------------------------------------------------------------------------
static inline void launch_gemm(const unsigned short* A, int lda,
                               const unsigned short* B, int ldb,
                               float* C, int ldc, int M, int N, int Nstore, int K,
                               float alpha, const float* bias, const float* colScale,
                               const float* addm, int ldadd, hipStream_t stream) {
  const int mtiles = M / 16;
  dim3 grid(N / 16, (mtiles + 3) / 4);
  k_gemm_bf16<<<grid, 128, 0, stream>>>(A, lda, B, ldb, C, ldc, mtiles, Nstore, K,
                                        alpha, bias, colScale, addm, ldadd);
}

extern "C" void kernel_launch(void* const* d_in, const int* in_sizes, int n_in,
                              void* d_out, int out_size, void* d_ws, size_t ws_size,
                              hipStream_t stream) {
  (void)in_sizes; (void)n_in; (void)out_size; (void)ws_size;
  const int*   x     = (const int*)  d_in[0];
  const float* lang  = (const float*)d_in[1];
  const float* img   = (const float*)d_in[2];
  const float* projW = (const float*)d_in[3];
  const float* projB = (const float*)d_in[4];
  const float* ln1g  = (const float*)d_in[5];
  const float* ln1b  = (const float*)d_in[6];
  const float* plog  = (const float*)d_in[7];
  const float* wsrc[7] = { (const float*)d_in[8],  (const float*)d_in[9],
                           (const float*)d_in[10], (const float*)d_in[11],
                           (const float*)d_in[14], (const float*)d_in[15],
                           (const float*)d_in[16] }; // inWr,inWi,outWr,outWi,gate,w1,w2
  const float* ln2g  = (const float*)d_in[12];
  const float* ln2b  = (const float*)d_in[13];
  float* out = (float*)d_out;

  // ---- workspace layout ----
  char* wp = (char*)d_ws;
  auto take = [&](size_t bytes) -> char* {
    char* p = wp; wp += (bytes + 255) & ~(size_t)255; return p;
  };
  unsigned short* tab_bf   = (unsigned short*)take(SZ_TAB * 2);
  unsigned short* projW_bf = (unsigned short*)take(SZ_PROJW * 2);
  float*          Wemb     = (float*)take(SZ_WEMB * 4);
  unsigned short* Wemb_bf  = (unsigned short*)take(SZ_WEMB * 2);
  unsigned short* wbf[7];
  for (int i = 0; i < 7; ++i) wbf[i] = (unsigned short*)take(SZ_W * 2);
  float* pparams = (float*)take((size_t)kNB * 3 * kD * 4);
  float* h   = (float*)take(SZ_ACT * 4);
  float* hid = (float*)take(SZ_ACT * 4);
  float* hcr = (float*)take(SZ_ACT * 4);
  float* hci = (float*)take(SZ_ACT * 4);
  float* tmp = (float*)take(SZ_ACT * 4);
  float* x1b = (float*)take(SZ_ACT * 4);
  unsigned short* abf   = (unsigned short*)take(SZ_ACT * 2);
  unsigned short* xo_bf = (unsigned short*)take(SZ_XO * 2);
  int* mask = (int*)take((size_t)kB * kLp * 4);

  auto ew = [](size_t n) { return dim3((unsigned)((n + 255) / 256)); };

  // ---- prep: bf16 tables/weights, exp(params) ----
  k_build_tables<<<ew(SZ_TAB), 256, 0, stream>>>(lang, img, tab_bf);
  k_f32_to_bf16<<<ew(SZ_PROJW), 256, 0, stream>>>(projW, projW_bf, SZ_PROJW);
  for (int i = 0; i < 7; ++i)
    k_f32_to_bf16<<<ew(SZ_W), 256, 0, stream>>>(wsrc[i], wbf[i], SZ_W);
  k_exp<<<ew((size_t)kNB * 3 * kD), 256, 0, stream>>>(plog, pparams, kNB * 3 * kD);

  // ---- W_emb = [lang|img] @ proj_W^T + proj_b  (8016x256, K=1280) ----
  launch_gemm(tab_bf, kDLI, projW_bf, kDLI, Wemb, kD, kVp, kD, kD, kDLI,
              1.0f, projB, nullptr, nullptr, 0, stream);
  k_f32_to_bf16<<<ew(SZ_WEMB), 256, 0, stream>>>(Wemb, Wemb_bf, SZ_WEMB);

  // ---- emb via row-gather of W_emb, front-padded; mask ----
  k_gather_embed<<<ew(SZ_ACT), 256, 0, stream>>>(x, Wemb, h, mask);

  const int Mact = kB * kLp;  // 4096 rows
  for (int blk = 0; blk < kNB; ++blk) {
    const float* nu    = pparams + (size_t)blk * 3 * kD;
    const float* theta = nu + kD;
    const float* gamma = nu + 2 * kD;
    const size_t wo = (size_t)blk * kD * kD;

    // LRU half: hid = MRL_LN1(h); hc = (hid @ (Wr+iWi)^T) * gamma
    k_mrl_ln<<<Mact, kD, 0, stream>>>(h, ln1g + blk * kD, ln1b + blk * kD, hid);
    k_f32_to_bf16<<<ew(SZ_ACT), 256, 0, stream>>>(hid, abf, SZ_ACT);
    launch_gemm(abf, kD, wbf[0] + wo, kD, hcr, kD, Mact, kD, kD, kD,
                1.0f, nullptr, gamma, nullptr, 0, stream);
    launch_gemm(abf, kD, wbf[1] + wo, kD, hci, kD, Mact, kD, kD, kD,
                1.0f, nullptr, gamma, nullptr, 0, stream);
    k_lru_scan<<<dim3(kD, kB), kLp, 0, stream>>>(hcr, hci, mask, nu, theta);
    // h = h + Re(hc @ (outWr+i outWi)^T) = h + hcr@Wr^T - hci@Wi^T
    k_f32_to_bf16<<<ew(SZ_ACT), 256, 0, stream>>>(hcr, abf, SZ_ACT);
    launch_gemm(abf, kD, wbf[2] + wo, kD, tmp, kD, Mact, kD, kD, kD,
                1.0f, nullptr, nullptr, h, kD, stream);
    k_f32_to_bf16<<<ew(SZ_ACT), 256, 0, stream>>>(hci, abf, SZ_ACT);
    launch_gemm(abf, kD, wbf[3] + wo, kD, h, kD, Mact, kD, kD, kD,
                -1.0f, nullptr, nullptr, tmp, kD, stream);

    // MLP half: h += silu(hid@gate^T) * (hid@w1^T) @ w2^T
    k_mrl_ln<<<Mact, kD, 0, stream>>>(h, ln2g + blk * kD, ln2b + blk * kD, hid);
    k_f32_to_bf16<<<ew(SZ_ACT), 256, 0, stream>>>(hid, abf, SZ_ACT);
    launch_gemm(abf, kD, wbf[4] + wo, kD, tmp, kD, Mact, kD, kD, kD,
                1.0f, nullptr, nullptr, nullptr, 0, stream);
    launch_gemm(abf, kD, wbf[5] + wo, kD, x1b, kD, Mact, kD, kD, kD,
                1.0f, nullptr, nullptr, nullptr, 0, stream);
    k_silu_mul<<<ew(SZ_ACT), 256, 0, stream>>>(tmp, x1b, hid, SZ_ACT);
    k_f32_to_bf16<<<ew(SZ_ACT), 256, 0, stream>>>(hid, abf, SZ_ACT);
    launch_gemm(abf, kD, wbf[6] + wo, kD, h, kD, Mact, kD, kD, kD,
                1.0f, nullptr, nullptr, h, kD, stream);
  }

  // ---- matryoshka scoring: scores_i = xo[..., :s] @ W_emb[:, :s]^T ----
  k_gather_xo<<<ew(SZ_XO), 256, 0, stream>>>(h, xo_bf);
  const int sizes[3] = {64, 128, 256};
  for (int i = 0; i < 3; ++i) {
    launch_gemm(xo_bf, kD, Wemb_bf, kD, out + (size_t)i * kB * kL0 * kV, kV,
                kB * kL0, kVp, kV, sizes[i],
                1.0f, nullptr, nullptr, nullptr, 0, stream);
  }
}
